// ResidualVQ_2826088481329
// MI455X (gfx1250) — matrix operand order repaired and checked by reading, not compiled, not run on hardware
//
#include <hip/hip_runtime.h>
#include <hip/hip_bf16.h>
#include <math.h>

// ---------------- problem constants ----------------
#define NQ      8
#define DIM     256
#define KCODES  1024
#define NTOK    65536          // 32 * 2048
#define MTILE   128            // tokens per workgroup
#define RPAD    260            // padded LDS row stride (words) -> conflict-free
#define THREADS 128            // 4 waves, each owns 32 tokens (2 WMMA A-tiles)

typedef __attribute__((ext_vector_type(16))) __bf16 v16bf;
typedef __attribute__((ext_vector_type(8)))  float  v8f;

// ---------------- prep kernels ----------------

// zero the atomic scratch (counts + loss sums)
__global__ void vq_init(float* __restrict__ counts, float* __restrict__ loss_sums) {
    int i = blockIdx.x * blockDim.x + threadIdx.x;
    if (i < NQ * KCODES) counts[i] = 0.0f;
    if (i < NQ)          loss_sums[i] = 0.0f;
}

// enorm[l][c] = sum_d embed[l][d][c]^2
__global__ void vq_enorm(const float* __restrict__ embeds, float* __restrict__ enorm) {
    int i = blockIdx.x * blockDim.x + threadIdx.x;      // 8192
    int l = i >> 10, c = i & (KCODES - 1);
    const float* e = embeds + (size_t)l * DIM * KCODES + c;
    float s = 0.0f;
    for (int d = 0; d < DIM; ++d) { float v = e[(size_t)d * KCODES]; s += v * v; }
    enorm[i] = s;
}

// et[l][c][d] = embed[l][d][c]   (f32 transposed codebook, contiguous code rows)
__global__ void vq_et(const float* __restrict__ embeds, float* __restrict__ et) {
    size_t i = (size_t)blockIdx.x * blockDim.x + threadIdx.x;   // 2^21 elems
    int d = (int)(i & (DIM - 1));
    int c = (int)((i >> 8) & (KCODES - 1));
    int l = (int)(i >> 18);
    et[i] = embeds[((size_t)l * DIM + d) * KCODES + c];
}

// Pre-swizzled bf16 B fragments: bfrag[l][t][s][lane][j] (j fastest, 16 bf16 per lane)
// B 32x16 k order used here (contiguous, same as A): lanes 0-15 -> N, K = s*32 + j ; lanes 16-31 -> N, K = s*32+16 + j
__global__ void vq_bfrag(const float* __restrict__ et, __bf16* __restrict__ bfrag) {
    size_t i = (size_t)blockIdx.x * blockDim.x + threadIdx.x;   // 2^21 elems
    int j    = (int)(i & 15);
    int lane = (int)((i >> 4) & 31);
    int s    = (int)((i >> 9) & 7);
    int t    = (int)((i >> 12) & 63);
    int l    = (int)(i >> 18);
    int n = t * 16 + (lane & 15);
    int k = s * 32 + (lane >> 4) * 16 + j;
    bfrag[i] = (__bf16)et[((size_t)l * KCODES + n) * DIM + k];
}

// ---------------- main residual-VQ kernel ----------------
__global__ __launch_bounds__(THREADS)
void vq_main(const float* __restrict__ x,
             const float* __restrict__ et,        // [NQ][K][D] f32
             const __bf16* __restrict__ bfrag,    // [NQ][64][8][32][16] bf16
             const float* __restrict__ enorm,     // [NQ][K]
             float* __restrict__ out,             // [NTOK][D]
             float* __restrict__ loss_sums,       // [NQ]
             float* __restrict__ counts)          // [NQ][K]
{
    __shared__ float r[MTILE][RPAD];
    __shared__ int   sel[MTILE];

    const int tid  = threadIdx.x;
    const int lane = tid & 31;
    const int wave = tid >> 5;                  // 0..3
    const int tok0 = blockIdx.x * MTILE;

    const int half  = lane >> 4;                // 0 / 1
    const int l16   = lane & 15;
    const int mytok = wave * 32;                // wave-local token base (2 A-tiles)

    // load x tile -> LDS residual (coalesced float4, padded rows)
    {
        const float4* xs = (const float4*)(x + (size_t)tok0 * DIM);
        for (int i = tid; i < MTILE * (DIM / 4); i += THREADS) {
            int row = i >> 6, col = i & 63;
            ((float4*)&r[row][0])[col] = xs[row * 64 + col];
        }
    }
    __syncthreads();

    for (int layer = 0; layer < NQ; ++layer) {
        // ---- build two A-tile fragment sets (-2 * residual, bf16) ----
        v16bf a0[8], a1[8];
        #pragma unroll
        for (int p = 0; p < 2; ++p) {
            const int tokA = mytok + p * 16 + l16;
            #pragma unroll
            for (int s = 0; s < 8; ++s) {
                const int kb = s * 32 + half * 16;
                const float4 lo0 = *(const float4*)&r[tokA][kb];
                const float4 lo1 = *(const float4*)&r[tokA][kb + 4];
                const float4 hi0 = *(const float4*)&r[tokA][kb + 8];
                const float4 hi1 = *(const float4*)&r[tokA][kb + 12];
                v16bf av;
                av[0]  = (__bf16)(-2.0f * lo0.x); av[1]  = (__bf16)(-2.0f * lo0.y);
                av[2]  = (__bf16)(-2.0f * lo0.z); av[3]  = (__bf16)(-2.0f * lo0.w);
                av[4]  = (__bf16)(-2.0f * lo1.x); av[5]  = (__bf16)(-2.0f * lo1.y);
                av[6]  = (__bf16)(-2.0f * lo1.z); av[7]  = (__bf16)(-2.0f * lo1.w);
                av[8]  = (__bf16)(-2.0f * hi0.x); av[9]  = (__bf16)(-2.0f * hi0.y);
                av[10] = (__bf16)(-2.0f * hi0.z); av[11] = (__bf16)(-2.0f * hi0.w);
                av[12] = (__bf16)(-2.0f * hi1.x); av[13] = (__bf16)(-2.0f * hi1.y);
                av[14] = (__bf16)(-2.0f * hi1.z); av[15] = (__bf16)(-2.0f * hi1.w);
                if (p == 0) a0[s] = av; else a1[s] = av;
            }
        }

        float b0v[8], b1v[8];
        int   b0i[8], b1i[8];
        #pragma unroll
        for (int i = 0; i < 8; ++i) {
            b0v[i] = 3.4e38f; b0i[i] = 0;
            b1v[i] = 3.4e38f; b1i[i] = 0;
        }

        const __bf16* bbase = bfrag + (size_t)layer * 64 * 8 * 512;
        const float*  enl   = enorm + layer * KCODES;

        // ---- sweep all 1024 codes in 64 N-tiles of 16; each B fragment feeds
        //      two WMMAs (two A-tiles) -> 2x arithmetic intensity on B ----
        float en_cur = enl[l16];
        for (int t = 0; t < 64; ++t) {
            const float en_next = enl[((t + 1) & 63) * 16 + l16];  // prefetched
            v8f acc0, acc1;
            #pragma unroll
            for (int i = 0; i < 8; ++i) { acc0[i] = en_cur; acc1[i] = en_cur; }

            const __bf16* bp = bbase + (size_t)t * 8 * 512 + lane * 16;
            #pragma unroll
            for (int s = 0; s < 8; ++s) {
                v16bf b = *(const v16bf*)(bp + s * 512);
                acc0 = __builtin_amdgcn_wmma_f32_16x16x32_bf16(
                        false, a0[s], false, b, (short)0, acc0, false, false);
                acc1 = __builtin_amdgcn_wmma_f32_16x16x32_bf16(
                        false, a1[s], false, b, (short)0, acc1, false, false);
            }
            const int code = t * 16 + l16;
            #pragma unroll
            for (int i = 0; i < 8; ++i) {
                if (acc0[i] < b0v[i]) { b0v[i] = acc0[i]; b0i[i] = code; }
                if (acc1[i] < b1v[i]) { b1v[i] = acc1[i]; b1i[i] = code; }
            }
            en_cur = en_next;
        }

        // ---- per-token argmin across the 16 lanes of each half-wave ----
        #pragma unroll
        for (int i = 0; i < 8; ++i) {
            float v = b0v[i]; int idx = b0i[i];
            #pragma unroll
            for (int m = 8; m >= 1; m >>= 1) {
                float ov = __shfl_xor(v, m, 16);
                int   oi = __shfl_xor(idx, m, 16);
                if (ov < v || (ov == v && oi < idx)) { v = ov; idx = oi; }
            }
            if (l16 == 0) sel[mytok + half * 8 + i] = idx;          // lanes 0 / 16
        }
        #pragma unroll
        for (int i = 0; i < 8; ++i) {
            float v = b1v[i]; int idx = b1i[i];
            #pragma unroll
            for (int m = 8; m >= 1; m >>= 1) {
                float ov = __shfl_xor(v, m, 16);
                int   oi = __shfl_xor(idx, m, 16);
                if (ov < v || (ov == v && oi < idx)) { v = ov; idx = oi; }
            }
            if (l16 == 0) sel[mytok + 16 + half * 8 + i] = idx;
        }
        __syncthreads();

        // ---- gather code (exact f32), update residual, accumulate loss/hist ----
        {
            const int code = sel[tid];                // one token per thread
            const float4* q4 = (const float4*)(et + ((size_t)layer * KCODES + code) * DIM);
            float4* r4 = (float4*)&r[tid][0];
            float lsum = 0.0f;
            #pragma unroll 8
            for (int d = 0; d < 64; ++d) {
                float4 qq = q4[d], rv = r4[d];
                float dx = qq.x - rv.x, dy = qq.y - rv.y;
                float dz = qq.z - rv.z, dw = qq.w - rv.w;
                lsum += dx * dx + dy * dy + dz * dz + dw * dw;
                r4[d] = make_float4(-dx, -dy, -dz, -dw);   // r_new = r - q
            }
            #pragma unroll
            for (int m = 16; m >= 1; m >>= 1) lsum += __shfl_xor(lsum, m, 32);
            if (lane == 0) atomicAdd(&loss_sums[layer], lsum);
            atomicAdd(&counts[layer * KCODES + code], 1.0f);
        }
        __syncthreads();
    }

    // ---- quantized_out = x - final_residual ----
    {
        const float4* xs = (const float4*)(x + (size_t)tok0 * DIM);
        float4* os = (float4*)(out + (size_t)tok0 * DIM);
        for (int i = tid; i < MTILE * (DIM / 4); i += THREADS) {
            int row = i >> 6, col = i & 63;
            float4 xv = xs[row * 64 + col];
            float4 rv = ((const float4*)&r[row][0])[col];
            os[row * 64 + col] = make_float4(xv.x - rv.x, xv.y - rv.y, xv.z - rv.z, xv.w - rv.w);
        }
    }
}

// ---------------- finalize: losses + perplexities ----------------
__global__ void vq_finalize(const float* __restrict__ loss_sums,
                            const float* __restrict__ counts,
                            float* __restrict__ out_losses,
                            float* __restrict__ out_perps)
{
    __shared__ float red[256];
    const int l = blockIdx.x, tid = threadIdx.x;
    float s = 0.0f;
    for (int c = tid; c < KCODES; c += 256) {
        float p = counts[l * KCODES + c] * (1.0f / (float)NTOK);
        s += p * logf(p + 1e-10f);
    }
    red[tid] = s; __syncthreads();
    for (int o = 128; o > 0; o >>= 1) { if (tid < o) red[tid] += red[tid + o]; __syncthreads(); }
    if (tid == 0) {
        out_perps[l]  = expf(-red[0]);
        out_losses[l] = loss_sums[l] * (1.0f / ((float)NTOK * (float)DIM));  // * COMMITMENT(1.0)
    }
}

// ---------------- launch ----------------
extern "C" void kernel_launch(void* const* d_in, const int* in_sizes, int n_in,
                              void* d_out, int out_size, void* d_ws, size_t ws_size,
                              hipStream_t stream) {
    (void)in_sizes; (void)n_in; (void)out_size; (void)ws_size;
    const float* x      = (const float*)d_in[0];   // [32,2048,256]
    const float* embeds = (const float*)d_in[1];   // [8,256,1024]
    float* out = (float*)d_out;                    // 16777216 + 8 + 8 floats

    // workspace carve-up (~12.6 MB)
    char* ws = (char*)d_ws;
    float*  et        = (float*)ws;                          ws += (size_t)NQ * KCODES * DIM * 4;   // 8 MB
    __bf16* bfrag     = (__bf16*)ws;                         ws += (size_t)NQ * 64 * 8 * 512 * 2;   // 4 MB
    float*  enorm     = (float*)ws;                          ws += (size_t)NQ * KCODES * 4;         // 32 KB
    float*  counts    = (float*)ws;                          ws += (size_t)NQ * KCODES * 4;         // 32 KB
    float*  loss_sums = (float*)ws;

    vq_init   <<<(NQ * KCODES + 255) / 256, 256, 0, stream>>>(counts, loss_sums);
    vq_enorm  <<<(NQ * KCODES) / 256,       256, 0, stream>>>(embeds, enorm);
    vq_et     <<<(NQ * KCODES * DIM) / 256, 256, 0, stream>>>(embeds, et);
    vq_bfrag  <<<(NQ * 64 * 8 * 512) / 256, 256, 0, stream>>>(et, bfrag);

    vq_main   <<<NTOK / MTILE, THREADS, 0, stream>>>(x, et, bfrag, enorm,
                                                     out, loss_sums, counts);

    vq_finalize<<<NQ, 256, 0, stream>>>(loss_sums, counts,
                                        out + (size_t)NTOK * DIM,
                                        out + (size_t)NTOK * DIM + NQ);
}
